// MultiHeadSelfAttention_18442589569536
// MI455X (gfx1250) — matrix operand order repaired and checked by reading, not compiled, not run on hardware
//
#include <hip/hip_runtime.h>
#include <hip/hip_bf16.h>
#include <stdint.h>

// ---------------- problem constants (from reference) ----------------
#define B_    2
#define T_    2048
#define D_    1024
#define H_    16
#define HS_   64
#define FEAT_ 1024

// ---------------- types ----------------
typedef __attribute__((ext_vector_type(16))) __bf16       v16bf;
typedef __attribute__((ext_vector_type(8)))  float        v8f;
typedef __attribute__((ext_vector_type(4)))  unsigned int u32x4;

union FragBF { v16bf v; u32x4 q[2]; };

__device__ __forceinline__ unsigned short f2bf_rn(float f) {
  unsigned int u = __float_as_uint(f);
  u += 0x7FFFu + ((u >> 16) & 1u);          // round-to-nearest-even
  return (unsigned short)(u >> 16);
}

// A-fragment (16x32 bf16, MxK), tile points at (row 0, k 0), row-major ld.
// Layout: lanes 0-15 M=lane, elems 0..15 -> K=0..15;
// lanes 16-31 add +16 to K.
__device__ __forceinline__ v16bf load_a_frag(const unsigned short* tile, int ld, int lane) {
  int m = lane & 15, half = lane >> 4;
  const unsigned short* p = tile + (size_t)m * ld + half * 16;
  FragBF f;
  f.q[0] = *(const u32x4*)(p);        // K = half*16 + (0..7)
  f.q[1] = *(const u32x4*)(p + 8);    // K = half*16 + 8 + (0..7)
  return f.v;
}

// B-fragment (32x16 bf16, KxN) from a TRANSPOSED operand stored [n][k] (ld in k).
// Layout: lane n = column, elems e -> K = half*16 + e (16 contiguous per lane).
__device__ __forceinline__ v16bf load_b_frag(const unsigned short* tileT, int ld, int lane) {
  int n = lane & 15, half = lane >> 4;
  const unsigned short* p = tileT + (size_t)n * ld + half * 16;
  FragBF f;
  f.q[0] = *(const u32x4*)(p);
  f.q[1] = *(const u32x4*)(p + 8);
  return f.v;
}

__device__ __forceinline__ v8f wmma_bf16(v16bf a, v16bf b, v8f c) {
  return __builtin_amdgcn_wmma_f32_16x16x32_bf16(false, a, false, b, (short)0, c, false, false);
}

// ---------------- prep kernels ----------------
__global__ void cvt_bf16_kernel(const float* __restrict__ in,
                                unsigned short* __restrict__ out, int n) {
  int i = blockIdx.x * blockDim.x + threadIdx.x;
  if (i < n) out[i] = f2bf_rn(in[i]);
}

// [H,D,HS] fp32 -> [H,HS,D] bf16
__global__ void wqkv_transpose_kernel(const float* __restrict__ w,
                                      unsigned short* __restrict__ wt) {
  int i = blockIdx.x * blockDim.x + threadIdx.x;      // output index h*HS*D + n*D + d
  int d = i % D_;
  int n = (i / D_) % HS_;
  int h = i / (D_ * HS_);
  wt[i] = f2bf_rn(w[((size_t)h * D_ + d) * HS_ + n]);
}

// [FEAT,FEAT] fp32 -> transposed [out][in] bf16
__global__ void wp_transpose_kernel(const float* __restrict__ w,
                                    unsigned short* __restrict__ wt) {
  int i = blockIdx.x * blockDim.x + threadIdx.x;      // o*FEAT + k
  int k = i % FEAT_;
  int o = i / FEAT_;
  wt[i] = f2bf_rn(w[(size_t)k * FEAT_ + o]);
}

// ---------------- fused QKV projection ----------------
// One wave -> 32x16 tiles of K, Q and V simultaneously:
//   2 A-fragments (x rows) shared by 3 weight B-fragments -> 6 WMMAs / k-step.
// k,q stored [B,H,T,HS] bf16 ; v stored transposed [B,H,HS,T] bf16.
__global__ __launch_bounds__(256) void qkv_proj_kernel(
    const unsigned short* __restrict__ xbf,
    const unsigned short* __restrict__ wkT,
    const unsigned short* __restrict__ wqT,
    const unsigned short* __restrict__ wvT,
    unsigned short* __restrict__ kbuf,
    unsigned short* __restrict__ qbuf,
    unsigned short* __restrict__ vTbuf) {
  int wave = (blockIdx.x * blockDim.x + threadIdx.x) >> 5;
  int lane = threadIdx.x & 31;
  int nt = wave & 3;                // HS/16
  int tt = (wave >> 2) & 63;        // T/32
  int h  = (wave >> 8) & 15;
  int b  = wave >> 12;

  const unsigned short* x0 = xbf + ((size_t)(b * T_ + tt * 32)) * D_;
  const unsigned short* x1 = x0 + (size_t)16 * D_;
  const unsigned short* wkt = wkT + ((size_t)(h * HS_ + nt * 16)) * D_;
  const unsigned short* wqt = wqT + ((size_t)(h * HS_ + nt * 16)) * D_;
  const unsigned short* wvt = wvT + ((size_t)(h * HS_ + nt * 16)) * D_;

  v8f ck[2] = {{}, {}}, cq[2] = {{}, {}}, cv[2] = {{}, {}};
  for (int k0 = 0; k0 < D_; k0 += 32) {
    v16bf a0 = load_a_frag(x0 + k0, D_, lane);
    v16bf a1 = load_a_frag(x1 + k0, D_, lane);
    v16bf bk = load_b_frag(wkt + k0, D_, lane);
    ck[0] = wmma_bf16(a0, bk, ck[0]);
    ck[1] = wmma_bf16(a1, bk, ck[1]);
    v16bf bq = load_b_frag(wqt + k0, D_, lane);
    cq[0] = wmma_bf16(a0, bq, cq[0]);
    cq[1] = wmma_bf16(a1, bq, cq[1]);
    v16bf bv = load_b_frag(wvt + k0, D_, lane);
    cv[0] = wmma_bf16(a0, bv, cv[0]);
    cv[1] = wmma_bf16(a1, bv, cv[1]);
  }

  int half = lane >> 4, n = lane & 15, m0 = half * 8;
  size_t bh = (size_t)(b * H_ + h);

  for (int tile = 0; tile < 2; ++tile) {
    int tbase = tt * 32 + tile * 16;
    unsigned short* kp = kbuf + (bh * T_ + tbase) * HS_ + nt * 16 + n;
    unsigned short* qp = qbuf + (bh * T_ + tbase) * HS_ + nt * 16 + n;
    for (int r = 0; r < 8; ++r) {
      kp[(size_t)(r + m0) * HS_] = f2bf_rn(ck[tile][r]);
      qp[(size_t)(r + m0) * HS_] = f2bf_rn(cq[tile][r]);
    }
    // vT: per lane the 8 values land on 8 consecutive t -> one b128 store
    unsigned int p0 = (unsigned)f2bf_rn(cv[tile][0]) | ((unsigned)f2bf_rn(cv[tile][1]) << 16);
    unsigned int p1 = (unsigned)f2bf_rn(cv[tile][2]) | ((unsigned)f2bf_rn(cv[tile][3]) << 16);
    unsigned int p2 = (unsigned)f2bf_rn(cv[tile][4]) | ((unsigned)f2bf_rn(cv[tile][5]) << 16);
    unsigned int p3 = (unsigned)f2bf_rn(cv[tile][6]) | ((unsigned)f2bf_rn(cv[tile][7]) << 16);
    u32x4 pk = {p0, p1, p2, p3};
    unsigned short* vp = vTbuf + (bh * HS_ + nt * 16 + n) * T_ + tbase + m0;
    *(u32x4*)vp = pk;
  }
}

// ---------------- flash attention ----------------
// Reference: wei[t,s] = k_t . q_s  -> k plays the query role, q plays the key role.
// One wave per (b, h, 16-query tile); query fragments live in registers for the
// whole key loop; online softmax over 32-key chunks; P re-laid out via LDS.
__global__ __launch_bounds__(256) void attn_kernel(
    const unsigned short* __restrict__ kbuf,   // query role [B,H,T,HS]
    const unsigned short* __restrict__ qbuf,   // key   role [B,H,T,HS]
    const unsigned short* __restrict__ vTbuf,  // [B,H,HS,T]
    unsigned short* __restrict__ att) {        // [B,T,FEAT] bf16 (heads concat)
  __shared__ unsigned short lds[8][16 * 32];   // per-wave P staging (1KB each)

  int wid  = threadIdx.x >> 5;
  int lane = threadIdx.x & 31;
  int wave = blockIdx.x * 8 + wid;
  int tt = wave & 127;
  int h  = (wave >> 7) & 15;
  int b  = wave >> 11;

  size_t bh = (size_t)(b * H_ + h);
  const unsigned short* kq = kbuf + (bh * T_ + (size_t)tt * 16) * HS_;
  const unsigned short* qk = qbuf + bh * T_ * HS_;
  const unsigned short* vt = vTbuf + bh * HS_ * T_;

  int half = lane >> 4, n = lane & 15, m0 = half * 8;
  int t0 = tt * 16;

  // Query fragments (16x64 split into two 16x32 K-chunks), reused for all keys.
  v16bf a0 = load_a_frag(kq, HS_, lane);
  v16bf a1 = load_a_frag(kq + 32, HS_, lane);

  v8f o0 = {}, o1 = {}, o2 = {}, o3 = {};
  float rmax[8], rsum[8];
  for (int r = 0; r < 8; ++r) { rmax[r] = -3.0e38f; rsum[r] = 0.f; }

  unsigned short* myl = lds[wid];
  int cmax = (t0 + 15) >> 5;                   // inclusive 32-key chunks

  for (int c = 0; c <= cmax; ++c) {
    int s0 = c * 32;
    v8f s[2];
    for (int sh = 0; sh < 2; ++sh) {
      const unsigned short* qrow = qk + (size_t)(s0 + sh * 16 + n) * HS_;
      FragBF f0, f1;                            // B[d][s]: lane n = key, contiguous d
      f0.q[0] = *(const u32x4*)(qrow + half * 16);
      f0.q[1] = *(const u32x4*)(qrow + half * 16 + 8);
      f1.q[0] = *(const u32x4*)(qrow + 32 + half * 16);
      f1.q[1] = *(const u32x4*)(qrow + 32 + half * 16 + 8);
      v8f acc = {};
      acc = wmma_bf16(a0, f0.v, acc);
      acc = wmma_bf16(a1, f1.v, acc);
      s[sh] = acc;
    }

    float alpha[8];
    for (int r = 0; r < 8; ++r) {
      int t = t0 + r + m0;
      float x0 = s[0][r] * 0.125f;              // HS^-0.5
      float x1 = s[1][r] * 0.125f;
      if (s0 + n > t)      x0 = -3.0e38f;       // causal mask
      if (s0 + 16 + n > t) x1 = -3.0e38f;
      float mx = fmaxf(x0, x1);
      for (int d = 1; d < 16; d <<= 1) mx = fmaxf(mx, __shfl_xor(mx, d, 32));
      float mnew = fmaxf(rmax[r], mx);
      float al = __expf(rmax[r] - mnew);
      rmax[r] = mnew;
      alpha[r] = al;
      float p0 = __expf(x0 - mnew);
      float p1 = __expf(x1 - mnew);
      float ps = p0 + p1;
      for (int d = 1; d < 16; d <<= 1) ps += __shfl_xor(ps, d, 32);
      rsum[r] = rsum[r] * al + ps;
      myl[(r + m0) * 32 + n]      = f2bf_rn(p0);   // P row-major 16x32 in LDS
      myl[(r + m0) * 32 + 16 + n] = f2bf_rn(p1);
    }
    for (int r = 0; r < 8; ++r) {
      o0[r] *= alpha[r]; o1[r] *= alpha[r]; o2[r] *= alpha[r]; o3[r] *= alpha[r];
    }

    // Re-layout P as an A fragment (16x32) via LDS, then out += P @ V.
    int m = lane & 15;
    FragBF pa;
    pa.q[0] = *(const u32x4*)(myl + m * 32 + half * 16);
    pa.q[1] = *(const u32x4*)(myl + m * 32 + half * 16 + 8);

    {
      const unsigned short* vr = vt + (size_t)(0 * 16 + n) * T_ + s0 + half * 16;
      FragBF vb; vb.q[0] = *(const u32x4*)vr; vb.q[1] = *(const u32x4*)(vr + 8);
      o0 = wmma_bf16(pa.v, vb.v, o0);
    }
    {
      const unsigned short* vr = vt + (size_t)(1 * 16 + n) * T_ + s0 + half * 16;
      FragBF vb; vb.q[0] = *(const u32x4*)vr; vb.q[1] = *(const u32x4*)(vr + 8);
      o1 = wmma_bf16(pa.v, vb.v, o1);
    }
    {
      const unsigned short* vr = vt + (size_t)(2 * 16 + n) * T_ + s0 + half * 16;
      FragBF vb; vb.q[0] = *(const u32x4*)vr; vb.q[1] = *(const u32x4*)(vr + 8);
      o2 = wmma_bf16(pa.v, vb.v, o2);
    }
    {
      const unsigned short* vr = vt + (size_t)(3 * 16 + n) * T_ + s0 + half * 16;
      FragBF vb; vb.q[0] = *(const u32x4*)vr; vb.q[1] = *(const u32x4*)(vr + 8);
      o3 = wmma_bf16(pa.v, vb.v, o3);
    }
  }

  // Normalize and store concat-head output tile [16 x 64] as bf16.
  unsigned short* ap = att + ((size_t)(b * T_ + t0)) * FEAT_ + h * HS_ + n;
  for (int r = 0; r < 8; ++r) {
    float inv = 1.0f / rsum[r];
    size_t row = (size_t)(r + m0) * FEAT_;
    ap[row + 0]  = f2bf_rn(o0[r] * inv);
    ap[row + 16] = f2bf_rn(o1[r] * inv);
    ap[row + 32] = f2bf_rn(o2[r] * inv);
    ap[row + 48] = f2bf_rn(o3[r] * inv);
  }
}

// ---------------- output projection ----------------
// One wave -> 32x32 output tile: 2 A-frags x 2 B-frags -> 4 WMMAs per k-step
// (16 FLOP per fragment byte, 2x reuse of both operands).
__global__ __launch_bounds__(256) void out_proj_kernel(
    const unsigned short* __restrict__ att,   // [B*T, FEAT] bf16
    const unsigned short* __restrict__ wpT,   // [FEAT(out), FEAT(in)] bf16
    const float* __restrict__ bp,
    float* __restrict__ out) {                // [B*T, FEAT] f32
  int wave = (blockIdx.x * blockDim.x + threadIdx.x) >> 5;
  int lane = threadIdx.x & 31;
  int nt = wave & 31;          // FEAT/32
  int mt = wave >> 5;          // (B*T)/32

  const unsigned short* a0t = att + (size_t)(mt * 32) * FEAT_;
  const unsigned short* a1t = a0t + (size_t)16 * FEAT_;
  const unsigned short* w0t = wpT + (size_t)(nt * 32) * FEAT_;
  const unsigned short* w1t = w0t + (size_t)16 * FEAT_;

  v8f c00 = {}, c01 = {}, c10 = {}, c11 = {};
  for (int k0 = 0; k0 < FEAT_; k0 += 32) {
    v16bf a0 = load_a_frag(a0t + k0, FEAT_, lane);
    v16bf a1 = load_a_frag(a1t + k0, FEAT_, lane);
    v16bf b0 = load_b_frag(w0t + k0, FEAT_, lane);
    v16bf b1 = load_b_frag(w1t + k0, FEAT_, lane);
    c00 = wmma_bf16(a0, b0, c00);
    c01 = wmma_bf16(a0, b1, c01);
    c10 = wmma_bf16(a1, b0, c10);
    c11 = wmma_bf16(a1, b1, c11);
  }

  int half = lane >> 4, n = lane & 15, m0 = half * 8;
  float bias0 = bp[nt * 32 + n];
  float bias1 = bp[nt * 32 + 16 + n];
  float* op = out + (size_t)(mt * 32) * FEAT_ + nt * 32 + n;
  for (int r = 0; r < 8; ++r) {
    size_t row0 = (size_t)(r + m0) * FEAT_;
    size_t row1 = (size_t)(r + m0 + 16) * FEAT_;
    op[row0 + 0]  = c00[r] + bias0;
    op[row0 + 16] = c01[r] + bias1;
    op[row1 + 0]  = c10[r] + bias0;
    op[row1 + 16] = c11[r] + bias1;
  }
}

// ---------------- host launcher ----------------
extern "C" void kernel_launch(void* const* d_in, const int* in_sizes, int n_in,
                              void* d_out, int out_size, void* d_ws, size_t ws_size,
                              hipStream_t stream) {
  (void)in_sizes; (void)n_in; (void)out_size; (void)ws_size;

  const float* x  = (const float*)d_in[0];
  const float* Wk = (const float*)d_in[1];
  const float* Wq = (const float*)d_in[2];
  const float* Wv = (const float*)d_in[3];
  const float* Wp = (const float*)d_in[4];
  const float* bp = (const float*)d_in[5];
  float* out = (float*)d_out;

  size_t off = 0;
  auto carve = [&](size_t bytes) -> void* {
    void* p = (char*)d_ws + off;
    off += (bytes + 255) & ~(size_t)255;
    return p;
  };
  unsigned short* xbf = (unsigned short*)carve((size_t)B_ * T_ * D_ * 2);
  unsigned short* wkT = (unsigned short*)carve((size_t)H_ * HS_ * D_ * 2);
  unsigned short* wqT = (unsigned short*)carve((size_t)H_ * HS_ * D_ * 2);
  unsigned short* wvT = (unsigned short*)carve((size_t)H_ * HS_ * D_ * 2);
  unsigned short* wpT = (unsigned short*)carve((size_t)FEAT_ * FEAT_ * 2);
  unsigned short* kbf = (unsigned short*)carve((size_t)B_ * H_ * T_ * HS_ * 2);
  unsigned short* qbf = (unsigned short*)carve((size_t)B_ * H_ * T_ * HS_ * 2);
  unsigned short* vT  = (unsigned short*)carve((size_t)B_ * H_ * HS_ * T_ * 2);
  unsigned short* atb = (unsigned short*)carve((size_t)B_ * T_ * FEAT_ * 2);

  // stage 0: precision/layout prep
  {
    int n = B_ * T_ * D_;
    cvt_bf16_kernel<<<n / 256, 256, 0, stream>>>(x, xbf, n);
  }
  {
    int n = H_ * HS_ * D_;
    wqkv_transpose_kernel<<<n / 256, 256, 0, stream>>>(Wk, wkT);
    wqkv_transpose_kernel<<<n / 256, 256, 0, stream>>>(Wq, wqT);
    wqkv_transpose_kernel<<<n / 256, 256, 0, stream>>>(Wv, wvT);
  }
  {
    int n = FEAT_ * FEAT_;
    wp_transpose_kernel<<<n / 256, 256, 0, stream>>>(Wp, wpT);
  }

  // stage 1: fused QKV projection  (B*H*(T/32)*(HS/16) = 8192 waves, 32x16 tiles x3)
  qkv_proj_kernel<<<8192 / 8, 256, 0, stream>>>(xbf, wkT, wqT, wvT, kbf, qbf, vT);

  // stage 2: flash attention       (B*H*(T/16) = 4096 waves)
  attn_kernel<<<4096 / 8, 256, 0, stream>>>(kbf, qbf, vT, atb);

  // stage 3: output projection     ((B*T/32)*(FEAT/32) = 4096 waves, 32x32 tiles)
  out_proj_kernel<<<4096 / 8, 256, 0, stream>>>(atb, wpT, bp, out);
}
